// Encoder_36850819400314
// MI455X (gfx1250) — compile-verified, matrix operand order checked by reading, not run on hardware
//
#include <hip/hip_runtime.h>
#include <hip/hip_bf16.h>
#include <hip/hip_fp16.h>

typedef __attribute__((ext_vector_type(16))) _Float16 v16h;
typedef __attribute__((ext_vector_type(8)))  _Float16 v8h;
typedef __attribute__((ext_vector_type(8)))  float    v8f;
typedef __attribute__((ext_vector_type(4)))  unsigned int u32x4;
typedef __attribute__((ext_vector_type(8)))  int      i32x8;
typedef __attribute__((ext_vector_type(4)))  int      i32x4;

#define DIM 128
#define NFEAT 78
#define KPAD 96
#define NLAYERS 5
#define BN_EPS 1e-5f

// ---------- fragment helpers ----------
__device__ __forceinline__ v16h make_frag(const _Float16* p0, const _Float16* p1) {
    v8h lo = *(const v8h*)p0;
    v8h hi = *(const v8h*)p1;
    v16h r;
#pragma unroll
    for (int i = 0; i < 8; ++i) { r[i] = lo[i]; r[i + 8] = hi[i]; }
    return r;
}

// A-matrix 16xK fragment from an LDS tile (row-major, row stride KT halfs).
__device__ __forceinline__ v16h load_a(const _Float16* tile, int KT, int mtile, int kc, int lane) {
    int m = mtile * 16 + (lane & 15);
    int k = kc * 32 + ((lane >> 4) << 3);
    const _Float16* base = tile + m * KT + k;
    return make_frag(base, base + 16);
}

// B-matrix Kx16 fragment from global transposed weights Wt[N][K] (row stride K halfs).
__device__ __forceinline__ v16h load_b(const _Float16* wt, int K, int ntile, int kc, int lane) {
    int n = ntile * 16 + (lane & 15);
    int k = kc * 32 + ((lane >> 4) << 3);
    const _Float16* base = wt + n * K + k;
    return make_frag(base, base + 16);
}

// ---------- Tensor Data Mover: contiguous 1D f16 tile, global -> LDS ----------
// D# per CDNA5 ISA ch.8: group0 {count=1, lds_addr, global_addr, type=2},
// group1 {data_size=2B, tensor_dim0=tile_dim0=nElems, dims1=1}, groups 2/3 zero (<=2D).
__device__ __forceinline__ void tdm_load_tile_f16(const _Float16* gsrc, unsigned lds_addr,
                                                  unsigned nElems) {
    unsigned long long ga = (unsigned long long)(uintptr_t)gsrc;
    u32x4 g0;
    g0[0] = 1u;                                     // count=1, is_restore=0, gather off
    g0[1] = lds_addr;                               // LDS byte address
    g0[2] = (unsigned)(ga & 0xFFFFFFFFu);           // global_addr[31:0]
    g0[3] = (unsigned)((ga >> 32) & 0x01FFFFFFu)    // global_addr[56:32]
          | (2u << 30);                             // type=2 ("image")
    unsigned long long d0 = nElems;                 // tensor_dim0
    unsigned long long st0 = nElems;                // tensor_dim0_stride
    union { unsigned long long q[4]; i32x8 v; } g1;
    g1.q[0] = (1ull << 16)                          // data_size=1 -> 2 bytes
            | ((d0 & 0xFFFFull) << 48);             // tensor_dim0[15:0]
    g1.q[1] = ((d0 >> 16) & 0xFFFFull)              // tensor_dim0[31:16]
            | (1ull << 16)                          // tensor_dim1 = 1
            | ((d0 & 0xFFFFull) << 48);             // tile_dim0 (= nElems, <=65535)
    g1.q[2] = 1ull                                  // tile_dim1 = 1
            | ((st0 & 0xFFFFFFFFull) << 32);        // tensor_dim0_stride[31:0]
    g1.q[3] = ((st0 >> 32) & 0xFFFFull)             // tensor_dim0_stride[47:32]
            | (st0 << 16);                          // tensor_dim1_stride
    i32x4 z4 = {};
#if __clang_major__ >= 23
    i32x8 z8 = {};
    __builtin_amdgcn_tensor_load_to_lds(g0, g1.v, z4, z4, z8, 0);
#else
    __builtin_amdgcn_tensor_load_to_lds(g0, g1.v, z4, z4, 0);
#endif
}

// ---------- small utility kernels ----------
__global__ void zero_kernel(float* p, size_t n) {
    size_t i = (size_t)blockIdx.x * blockDim.x + threadIdx.x;
    if (i < n) p[i] = 0.f;
}

// Wt[n*Kpad + k] = (k<K) ? W[k*N + n] : 0   (fp32 -> f16, transpose, K-pad)
__global__ void convert_wt_kernel(const float* __restrict__ W, _Float16* __restrict__ Wt,
                                  int K, int N, int Kpad) {
    int i = blockIdx.x * blockDim.x + threadIdx.x;
    if (i >= N * Kpad) return;
    int n = i / Kpad, k = i % Kpad;
    float v = (k < K) ? W[k * N + n] : 0.f;
    Wt[n * Kpad + k] = (_Float16)v;
}

// ---------- ini_embed: h16 = relu(x@W1+b1)@W2+b2 ----------
__global__ __launch_bounds__(256) void ini_embed_kernel(
    const float* __restrict__ x, const _Float16* __restrict__ wt1, const float* __restrict__ b1,
    const _Float16* __restrict__ wt2, const float* __restrict__ b2,
    _Float16* __restrict__ h16, int nNodes)
{
    __shared__ __align__(16) _Float16 lds[DIM * DIM];
    const int tid  = threadIdx.x;
    const int lane = tid & 31;
    const int wave = tid >> 5;
    const int node0 = blockIdx.x * 128;

    // stage x tile [128 x 96] (pad 78->96) as f16
    for (int i = tid; i < 128 * KPAD; i += 256) {
        int m = i / KPAD, k = i - m * KPAD;
        int node = node0 + m;
        float v = (k < NFEAT && node < nNodes) ? x[(size_t)node * NFEAT + k] : 0.f;
        lds[m * KPAD + k] = (_Float16)v;
    }
    __syncthreads();

    v8f zero = {};
    v8f acc[8];
#pragma unroll
    for (int nt = 0; nt < 8; ++nt) acc[nt] = zero;

    // GEMM1: [128x96] x [96x128]
#pragma unroll
    for (int kc = 0; kc < 3; ++kc) {
        v16h a = load_a(lds, KPAD, wave, kc, lane);
#pragma unroll
        for (int nt = 0; nt < 8; ++nt) {
            v16h b = load_b(wt1, KPAD, nt, kc, lane);
            acc[nt] = __builtin_amdgcn_wmma_f32_16x16x32_f16(false, a, false, b,
                                                             (short)0, acc[nt], false, false);
        }
    }
    __syncthreads();

    const int nbase = lane & 15;
    const int moff  = (lane >> 4) << 3;
#pragma unroll
    for (int nt = 0; nt < 8; ++nt) {
        int n = nt * 16 + nbase;
        float bias = b1[n];
#pragma unroll
        for (int r = 0; r < 8; ++r) {
            int m = wave * 16 + moff + r;
            float v = acc[nt][r] + bias;
            lds[m * DIM + n] = (_Float16)(v > 0.f ? v : 0.f);
        }
    }
    __syncthreads();

    // GEMM2: [128x128] x [128x128]
    v8f acc2[8];
#pragma unroll
    for (int nt = 0; nt < 8; ++nt) acc2[nt] = zero;
#pragma unroll
    for (int kc = 0; kc < 4; ++kc) {
        v16h a = load_a(lds, DIM, wave, kc, lane);
#pragma unroll
        for (int nt = 0; nt < 8; ++nt) {
            v16h b = load_b(wt2, DIM, nt, kc, lane);
            acc2[nt] = __builtin_amdgcn_wmma_f32_16x16x32_f16(false, a, false, b,
                                                              (short)0, acc2[nt], false, false);
        }
    }
#pragma unroll
    for (int nt = 0; nt < 8; ++nt) {
        int n = nt * 16 + nbase;
        float bias = b2[n];
#pragma unroll
        for (int r = 0; r < 8; ++r) {
            int m = wave * 16 + moff + r;
            int node = node0 + m;
            if (node < nNodes) h16[(size_t)node * DIM + n] = (_Float16)(acc2[nt][r] + bias);
        }
    }
}

// ---------- edge scatter: agg16[dst] += h16[src], packed-f16 atomics, one wave/edge ----------
__global__ __launch_bounds__(256) void scatter_add_kernel(
    const _Float16* __restrict__ h16, const int* __restrict__ src, const int* __restrict__ dst,
    _Float16* __restrict__ agg16, int nEdges)
{
    int gid  = blockIdx.x * blockDim.x + threadIdx.x;
    int e    = gid >> 5;
    int lane = gid & 31;
    if (e >= nEdges) return;
    int s = src[e], d = dst[e];
    const __half2* hp = (const __half2*)(h16 + (size_t)s * DIM) + lane * 2;
    __half2 v0 = hp[0];
    __half2 v1 = hp[1];
    __half2* p = (__half2*)(agg16 + (size_t)d * DIM) + lane * 2;
    atomicAdd(p + 0, v0);   // global_atomic_pk_add_f16
    atomicAdd(p + 1, v1);
}

// ---------- GIN MLP (in place on agg16) + BN statistics; TDM-staged input ----------
__global__ __launch_bounds__(256) void gin_mlp_kernel(
    _Float16* __restrict__ agg16, const _Float16* __restrict__ wt1, const float* __restrict__ b1,
    const _Float16* __restrict__ wt2, const float* __restrict__ b2,
    float* __restrict__ gsum, float* __restrict__ gsumsq, int nNodes)
{
    __shared__ __align__(16) _Float16 lds[DIM * DIM];
    __shared__ float ssum[DIM];
    __shared__ float ssq[DIM];
    const int tid  = threadIdx.x;
    const int lane = tid & 31;
    const int wave = tid >> 5;
    const int node0 = blockIdx.x * 128;

    // Tensor Data Mover: 32KB contiguous f16 tile -> LDS (rows past nNodes are
    // garbage but their outputs are masked below).
    if (tid == 0) {
        tdm_load_tile_f16(agg16 + (size_t)node0 * DIM, (unsigned)(uintptr_t)lds, 128 * DIM);
        __builtin_amdgcn_s_wait_tensorcnt(0);
    }
    if (tid < DIM) { ssum[tid] = 0.f; ssq[tid] = 0.f; }
    __syncthreads();

    v8f zero = {};
    v8f acc[8];
#pragma unroll
    for (int nt = 0; nt < 8; ++nt) acc[nt] = zero;

    // GEMM1
#pragma unroll
    for (int kc = 0; kc < 4; ++kc) {
        v16h a = load_a(lds, DIM, wave, kc, lane);
#pragma unroll
        for (int nt = 0; nt < 8; ++nt) {
            v16h b = load_b(wt1, DIM, nt, kc, lane);
            acc[nt] = __builtin_amdgcn_wmma_f32_16x16x32_f16(false, a, false, b,
                                                             (short)0, acc[nt], false, false);
        }
    }
    __syncthreads();

    const int nbase = lane & 15;
    const int moff  = (lane >> 4) << 3;
#pragma unroll
    for (int nt = 0; nt < 8; ++nt) {
        int n = nt * 16 + nbase;
        float bias = b1[n];
#pragma unroll
        for (int r = 0; r < 8; ++r) {
            int m = wave * 16 + moff + r;
            float v = acc[nt][r] + bias;
            lds[m * DIM + n] = (_Float16)(v > 0.f ? v : 0.f);
        }
    }
    __syncthreads();

    // GEMM2
    v8f acc2[8];
#pragma unroll
    for (int nt = 0; nt < 8; ++nt) acc2[nt] = zero;
#pragma unroll
    for (int kc = 0; kc < 4; ++kc) {
        v16h a = load_a(lds, DIM, wave, kc, lane);
#pragma unroll
        for (int nt = 0; nt < 8; ++nt) {
            v16h b = load_b(wt2, DIM, nt, kc, lane);
            acc2[nt] = __builtin_amdgcn_wmma_f32_16x16x32_f16(false, a, false, b,
                                                              (short)0, acc2[nt], false, false);
        }
    }

    // bias + final relu, store z (f16) in place, accumulate BN stats in LDS
#pragma unroll
    for (int nt = 0; nt < 8; ++nt) {
        int n = nt * 16 + nbase;
        float bias = b2[n];
#pragma unroll
        for (int r = 0; r < 8; ++r) {
            int m = wave * 16 + moff + r;
            int node = node0 + m;
            if (node < nNodes) {
                float z = acc2[nt][r] + bias;
                z = z > 0.f ? z : 0.f;
                agg16[(size_t)node * DIM + n] = (_Float16)z;
                atomicAdd(&ssum[n], z);
                atomicAdd(&ssq[n], z * z);
            }
        }
    }
    __syncthreads();
    if (tid < DIM) {
        atomicAdd(&gsum[tid], ssum[tid]);
        atomicAdd(&gsumsq[tid], ssq[tid]);
    }
}

// ---------- BN finalize (1 block, 128 threads) ----------
__global__ void bn_finalize_kernel(float* __restrict__ sums, float* __restrict__ sumsq,
                                   const float* __restrict__ gamma, const float* __restrict__ beta,
                                   float* __restrict__ scale, float* __restrict__ shift,
                                   int layer, float invN)
{
    int d = threadIdx.x;
    if (d >= DIM) return;
    float mean = sums[d] * invN;
    float var  = sumsq[d] * invN - mean * mean;
    float s = gamma[layer * DIM + d] * rsqrtf(var + BN_EPS);
    scale[d] = s;
    shift[d] = beta[layer * DIM + d] - mean * s;
    sums[d] = 0.f;   // reset for next layer
    sumsq[d] = 0.f;
}

// ---------- BN apply + h16 update + layer-conv accumulate (8 halfs / thread) ----------
__global__ __launch_bounds__(256) void bn_apply_kernel(
    const _Float16* __restrict__ z16, _Float16* __restrict__ h16, float* __restrict__ pos,
    const float* __restrict__ scale, const float* __restrict__ shift,
    const float* __restrict__ lcw, int layer, int nNodes)
{
    int i = blockIdx.x * blockDim.x + threadIdx.x;
    int node = i >> 4;
    if (node >= nNodes) return;
    int c = (i & 15) * 8;
    float w = lcw[layer];
    v8h zv = *(const v8h*)(z16 + (size_t)node * DIM + c);
    v8h hv;
    float f[8];
#pragma unroll
    for (int j = 0; j < 8; ++j) {
        float v = (float)zv[j] * scale[c + j] + shift[c + j];
        f[j] = v;
        hv[j] = (_Float16)v;
    }
    *(v8h*)(h16 + (size_t)node * DIM + c) = hv;
    float* pp = pos + (size_t)node * DIM + c;
    float4 p0 = *(float4*)(pp);
    float4 p1 = *(float4*)(pp + 4);
    p0.x += w * f[0]; p0.y += w * f[1]; p0.z += w * f[2]; p0.w += w * f[3];
    p1.x += w * f[4]; p1.y += w * f[5]; p1.z += w * f[6]; p1.w += w * f[7];
    *(float4*)(pp)     = p0;
    *(float4*)(pp + 4) = p1;
}

// ---------- global_add_pool over sorted batch ----------
__global__ __launch_bounds__(256) void pool_kernel(
    const float* __restrict__ pos, const int* __restrict__ batch,
    const float* __restrict__ lcb, float* __restrict__ out, int nNodes)
{
    int i = blockIdx.x * blockDim.x + threadIdx.x;
    int node = i >> 5;
    if (node >= nNodes) return;
    int c = (i & 31) * 4;
    int g = batch[node];
    float b = lcb[0];
    const float4 v = *(const float4*)(pos + (size_t)node * DIM + c);
    float* o = out + (size_t)g * DIM + c;
    atomicAdd(o + 0, v.x + b);
    atomicAdd(o + 1, v.y + b);
    atomicAdd(o + 2, v.z + b);
    atomicAdd(o + 3, v.w + b);
}

// ---------- host launcher ----------
extern "C" void kernel_launch(void* const* d_in, const int* in_sizes, int n_in,
                              void* d_out, int out_size, void* d_ws, size_t ws_size,
                              hipStream_t stream) {
    const float* x        = (const float*)d_in[0];
    const int*   eidx     = (const int*)d_in[1];
    const int*   batch    = (const int*)d_in[2];
    const float* ini_w1   = (const float*)d_in[4];
    const float* ini_b1   = (const float*)d_in[5];
    const float* ini_w2   = (const float*)d_in[6];
    const float* ini_b2   = (const float*)d_in[7];
    const float* gin_w1   = (const float*)d_in[8];
    const float* gin_b1   = (const float*)d_in[9];
    const float* gin_w2   = (const float*)d_in[10];
    const float* gin_b2   = (const float*)d_in[11];
    const float* bn_gamma = (const float*)d_in[12];
    const float* bn_beta  = (const float*)d_in[13];
    const float* lc_w     = (const float*)d_in[14];
    const float* lc_b     = (const float*)d_in[15];
    float* out = (float*)d_out;

    const int N = in_sizes[2];            // nodes
    const int E = in_sizes[1] / 2;        // edges
    const size_t NH = (size_t)N * DIM * sizeof(_Float16);   // f16 node tensor bytes
    const size_t NB = (size_t)N * DIM * sizeof(float);      // fp32 node tensor bytes

    char* base = (char*)d_ws;
    _Float16* h16   = (_Float16*)(base);
    _Float16* agg16 = (_Float16*)(base + NH);
    float*    pos   = (float*)(base + 2 * NH);
    _Float16* wt1   = (_Float16*)(base + 2 * NH + NB);                 // 128x96
    _Float16* wt2   = (_Float16*)(base + 2 * NH + NB + 24576);         // 128x128
    _Float16* gwt1  = (_Float16*)(base + 2 * NH + NB + 24576 + 32768); // 5 x 128x128
    _Float16* gwt2  = (_Float16*)(base + 2 * NH + NB + 24576 + 32768 + 163840);
    float*    stats = (float*)(base + 2 * NH + NB + 24576 + 32768 + 2 * 163840);
    float* sums  = stats;
    float* sumsq = stats + DIM;
    float* scale = stats + 2 * DIM;
    float* shift = stats + 3 * DIM;

    const int tiles = (N + 127) / 128;

    // zero accumulators / output
    zero_kernel<<<(unsigned)(((size_t)N * DIM + 255) / 256), 256, 0, stream>>>(pos, (size_t)N * DIM);
    zero_kernel<<<(unsigned)((out_size + 255) / 256), 256, 0, stream>>>(out, (size_t)out_size);
    zero_kernel<<<1, 256, 0, stream>>>(stats, 4 * DIM);

    // convert + transpose weights to f16 (N-major)
    convert_wt_kernel<<<(128 * KPAD + 255) / 256, 256, 0, stream>>>(ini_w1, wt1, NFEAT, DIM, KPAD);
    convert_wt_kernel<<<(DIM * DIM + 255) / 256, 256, 0, stream>>>(ini_w2, wt2, DIM, DIM, DIM);
    for (int l = 0; l < NLAYERS; ++l) {
        convert_wt_kernel<<<(DIM * DIM + 255) / 256, 256, 0, stream>>>(
            gin_w1 + (size_t)l * DIM * DIM, gwt1 + (size_t)l * DIM * DIM, DIM, DIM, DIM);
        convert_wt_kernel<<<(DIM * DIM + 255) / 256, 256, 0, stream>>>(
            gin_w2 + (size_t)l * DIM * DIM, gwt2 + (size_t)l * DIM * DIM, DIM, DIM, DIM);
    }

    // ini_embed
    ini_embed_kernel<<<tiles, 256, 0, stream>>>(x, wt1, ini_b1, wt2, ini_b2, h16, N);

    const float invN = 1.0f / (float)N;
    for (int l = 0; l < NLAYERS; ++l) {
        // agg = h  (self term of GINConv, eps=0)
        hipMemcpyAsync(agg16, h16, NH, hipMemcpyDeviceToDevice, stream);
        // agg[dst] += h[src]   (packed f16 atomics)
        scatter_add_kernel<<<(E * 32 + 255) / 256, 256, 0, stream>>>(
            h16, eidx + 0, eidx + E, agg16, E);
        // z = relu(mlp(agg)) in place; BN stats into sums/sumsq
        gin_mlp_kernel<<<tiles, 256, 0, stream>>>(
            agg16, gwt1 + (size_t)l * DIM * DIM, gin_b1 + (size_t)l * DIM,
            gwt2 + (size_t)l * DIM * DIM, gin_b2 + (size_t)l * DIM,
            sums, sumsq, N);
        bn_finalize_kernel<<<1, 128, 0, stream>>>(sums, sumsq, bn_gamma, bn_beta,
                                                  scale, shift, l, invN);
        // h = BN(z); pos += lc_w[l] * h
        bn_apply_kernel<<<(N * 16 + 255) / 256, 256, 0, stream>>>(
            agg16, h16, pos, scale, shift, lc_w, l, N);
    }

    // out[g] = sum_{node in g} (pos[node] + lc_b)
    pool_kernel<<<(N * 32 + 255) / 256, 256, 0, stream>>>(pos, batch, lc_b, out, N);
}
